// EmergentResonator_19353122636156
// MI455X (gfx1250) — compile-verified, matrix operand order checked by reading, not run on hardware
//
#include <hip/hip_runtime.h>
#include <hip/hip_bf16.h>
#include <stdint.h>

// ---------------- problem constants ----------------
constexpr int kB = 256, kT = 64, kI = 128, kN = 2048, kC = 10;
constexpr float kLEAK = 0.95f;
constexpr int kRSTEPS = 12;
constexpr int kNSTEPS = kT + kRSTEPS;          // 76
constexpr float kLN_EPS = 1e-5f;

// ---------------- tiling ----------------
constexpr int BTILE = 16;                      // batch rows per WG (one WMMA M tile)
constexpr int NTILE = 128;                     // neurons per WG (8 waves x 16)
constexpr int NBT   = kB / BTILE;              // 16 batch tiles
constexpr int NNS   = kN / NTILE;              // 16 N slices
constexpr int NWG   = NBT * NNS;               // 256 workgroups (persistent, grid-synced)
constexpr int KW    = 32;                      // bf16 WMMA K
constexpr int NKN   = kN / KW;                 // 64 K tiles over neurons
constexpr int NKI   = kI / KW;                 // 4  K tiles over inputs
constexpr int KCHUNK = 16;                     // K tiles staged per LDS chunk (16KB)
constexpr int NCHUNK = NKN / KCHUNK;           // 4 chunks per step

typedef __bf16 v16bf __attribute__((ext_vector_type(16)));
typedef float  v8f   __attribute__((ext_vector_type(8)));
typedef int    v4i   __attribute__((ext_vector_type(4)));

typedef __attribute__((address_space(1))) v4i* gas_v4i;   // global (AS1) v4i*
typedef __attribute__((address_space(3))) v4i* las_v4i;   // LDS (AS3) v4i*

#if __has_builtin(__builtin_amdgcn_global_load_async_to_lds_b128)
#define HAVE_ASYNC_LDS 1
#endif

__device__ __forceinline__ void copy16_to_lds(const uint32_t* g, uint32_t* l) {
#ifdef HAVE_ASYNC_LDS
  __builtin_amdgcn_global_load_async_to_lds_b128((gas_v4i)(void*)g, (las_v4i)(void*)l, 0, 0);
#else
  *(uint4*)l = *(const uint4*)g;
#endif
}

__device__ __forceinline__ void wait_async_copies() {
#ifdef HAVE_ASYNC_LDS
#if __has_builtin(__builtin_amdgcn_s_wait_asynccnt)
  __builtin_amdgcn_s_wait_asynccnt(0);
#else
  asm volatile("s_wait_asynccnt 0" ::: "memory");
#endif
#endif
}

// K index inside a 16x32 bf16 A fragment (ISA 7.12.2): VGPR p, lane-half h
// lanes 0-15: K = 0..7,16..23 ; lanes 16-31: K = 8..15,24..31 (pairs per dword)
__device__ __forceinline__ int kidx(int p, int h) {
  return (p < 4 ? 2 * p : 16 + 2 * (p - 4)) + 8 * h;
}

__device__ __forceinline__ uint32_t f2bf(float f) {
  uint32_t u = __float_as_uint(f);
  return ((u + 0x7FFFu + ((u >> 16) & 1u)) >> 16) & 0xFFFFu;  // RNE
}
__device__ __forceinline__ uint32_t packbf(float lo, float hi) {
  return f2bf(lo) | (f2bf(hi) << 16);
}

// ------------------------------------------------------------------
// One-time packing: W_rec -> B fragments, W_in -> B fragments,
// x -> A fragments, all bf16, fragment-major (contiguous 1KiB/fragment).
// ------------------------------------------------------------------
constexpr int NWF = NNS * 8 * NKN;   // 8192 W_rec fragments
constexpr int NWI = NNS * 8 * NKI;   // 512  W_in  fragments
constexpr int NXA = NBT * kT * NKI;  // 4096 x     fragments

__global__ void pack_kernel(const float* __restrict__ x,
                            const float* __restrict__ W_in,
                            const float* __restrict__ W_rec,
                            uint32_t* __restrict__ wpack,
                            uint32_t* __restrict__ winpack,
                            uint32_t* __restrict__ xapack) {
  int tid  = blockIdx.x * blockDim.x + threadIdx.x;
  int lane = tid & 31;
  int frag = tid >> 5;
  int hl = lane >> 4, l16 = lane & 15;
  uint32_t d[8];
  if (frag < NWF) {                                   // W_rec: B frag (k,n) = W_rec[n][k]
    int ntile = frag / NKN, ktile = frag % NKN;
    const float* row = W_rec + (size_t)(ntile * 16 + l16) * kN + ktile * KW;
#pragma unroll
    for (int p = 0; p < 8; p++) { int k = kidx(p, hl); d[p] = packbf(row[k], row[k + 1]); }
    uint32_t* dst = wpack + ((size_t)frag * 32 + lane) * 8;
    *(uint4*)dst = *(uint4*)d; *(uint4*)(dst + 4) = *(uint4*)(d + 4);
  } else if (frag < NWF + NWI) {                      // W_in: B frag
    int f = frag - NWF;
    int ntile = f / NKI, ktile = f % NKI;
    const float* row = W_in + (size_t)(ntile * 16 + l16) * kI + ktile * KW;
#pragma unroll
    for (int p = 0; p < 8; p++) { int k = kidx(p, hl); d[p] = packbf(row[k], row[k + 1]); }
    uint32_t* dst = winpack + ((size_t)f * 32 + lane) * 8;
    *(uint4*)dst = *(uint4*)d; *(uint4*)(dst + 4) = *(uint4*)(d + 4);
  } else if (frag < NWF + NWI + NXA) {                // x: A frag (row = batch, K = input dim)
    int f = frag - NWF - NWI;
    int bt = f / (kT * NKI), r = f % (kT * NKI);
    int t = r / NKI, ktile = r % NKI;
    const float* row = x + ((size_t)(bt * 16 + l16) * kT + t) * kI + ktile * KW;
#pragma unroll
    for (int p = 0; p < 8; p++) { int k = kidx(p, hl); d[p] = packbf(row[k], row[k + 1]); }
    uint32_t* dst = xapack + ((size_t)f * 32 + lane) * 8;
    *(uint4*)dst = *(uint4*)d; *(uint4*)(dst + 4) = *(uint4*)(d + 4);
  }
}

__global__ void zero_kernel(uint32_t* p, int n) {
  int i = blockIdx.x * blockDim.x + threadIdx.x;
  if (i < n) p[i] = 0u;
}

// one-shot grid barrier slot: [0]=arrive counter, [1]=release flag (pre-zeroed)
__device__ __forceinline__ void grid_barrier(uint32_t* slot, int tid) {
  __syncthreads();
  __threadfence();
  if (tid == 0) {
    if (__hip_atomic_fetch_add(slot, 1u, __ATOMIC_ACQ_REL, __HIP_MEMORY_SCOPE_AGENT) == NWG - 1) {
      __hip_atomic_store(slot + 1, 1u, __ATOMIC_RELEASE, __HIP_MEMORY_SCOPE_AGENT);
    } else {
      while (__hip_atomic_load(slot + 1, __ATOMIC_ACQUIRE, __HIP_MEMORY_SCOPE_AGENT) == 0u)
        __builtin_amdgcn_s_sleep(2);
    }
  }
  __syncthreads();
}

// ------------------------------------------------------------------
// Persistent fused recurrent kernel. WG (bt,ns): 16 batch rows x 128 neurons.
// Membrane potential lives in LDS for all 76 steps. Spikes double-buffered
// in global memory as pre-packed bf16 A fragments; A fragments are staged
// into LDS once per WG via the async copy engine (shared by all 8 waves),
// double-buffered in 16KB chunks so the DMA overlaps the WMMAs.
// ------------------------------------------------------------------
__global__ void __launch_bounds__(256, 1) resonator_kernel(
    const float* __restrict__ b_in, const float* __restrict__ gamma,
    const float* __restrict__ beta, const float* __restrict__ thr,
    const float* __restrict__ intr, const float* __restrict__ steep,
    const float* __restrict__ W_cls,
    const uint32_t* __restrict__ wpack, const uint32_t* __restrict__ winpack,
    const uint32_t* __restrict__ xapack,
    uint32_t* __restrict__ spk0, uint32_t* __restrict__ spk1,
    float* __restrict__ red, float* __restrict__ cls_part,
    uint32_t* __restrict__ bars) {
  __shared__ uint32_t ldsA[2][KCHUNK * 256];   // 2 x 16KB A-fragment staging
  __shared__ float cur[16][NTILE];
  __shared__ float pot[16][NTILE];
  __shared__ float spk[16][NTILE];
  __shared__ float mu_s[16], rs_s[16];

  const int tid  = threadIdx.x;
  const int bt   = blockIdx.x / NNS;
  const int ns   = blockIdx.x % NNS;
  const int wave = tid >> 5;
  const int lane = tid & 31;
  const int l16  = lane & 15, hl = lane >> 4;

  for (int i = tid; i < 16 * NTILE; i += 256) (&pot[0][0])[i] = 0.f;

  // per-wave output-column constants (D layout: col = lane%16, row = vgpr + 8*half)
  const int n16  = ns * 8 + wave;                 // global 16-wide N tile
  const float bi = b_in[n16 * 16 + l16];

  // per-thread pulse constants: thread -> (row = tid/16, 8 cols at (tid%16)*8)
  const int prow = tid >> 4, pcg = tid & 15;
  float g8[8], be8[8], th8[8], in8[8], sa8[8];
#pragma unroll
  for (int j = 0; j < 8; j++) {
    int n = ns * NTILE + pcg * 8 + j;
    g8[j] = gamma[n]; be8[j] = beta[n]; th8[j] = thr[n];
    in8[j] = intr[n]; sa8[j] = fabsf(steep[n]);
  }
  __syncthreads();

  for (int s = 0; s < kNSTEPS; ++s) {
    const uint32_t* rbuf = (s & 1) ? spk1 : spk0;
    uint32_t*       wbuf = (s & 1) ? spk0 : spk1;

    // ---------------- phase A: bf16 WMMA GEMM, f32 accumulate ----------------
    v8f c = {0.f, 0.f, 0.f, 0.f, 0.f, 0.f, 0.f, 0.f};
    {
      const uint32_t* achunk0 = rbuf + (size_t)bt * NKN * 256;   // this WG's A stream
      const uint32_t* bbase   = wpack + (size_t)n16 * NKN * 256 + lane * 8;

      // kick off async copy of chunk 0 (16KB, 4 x b128 per thread)
#pragma unroll
      for (int j = 0; j < 4; j++) {
        int u = tid + j * 256;                        // 16B unit within chunk
        copy16_to_lds(achunk0 + u * 4, &ldsA[0][u * 4]);
      }

#pragma unroll 1
      for (int q = 0; q < NCHUNK; ++q) {
        wait_async_copies();
        __syncthreads();                              // chunk q resident for all waves
        if (q + 1 < NCHUNK) {                         // prefetch chunk q+1 into other buffer
          const uint32_t* gsrc = achunk0 + (size_t)(q + 1) * KCHUNK * 256;
#pragma unroll
          for (int j = 0; j < 4; j++) {
            int u = tid + j * 256;
            copy16_to_lds(gsrc + u * 4, &ldsA[(q + 1) & 1][u * 4]);
          }
        }
        const uint32_t* aL = &ldsA[q & 1][lane * 8];
#pragma unroll 4
        for (int kt2 = 0; kt2 < KCHUNK; ++kt2) {
          union { uint4 u[2]; v16bf v; } fa, fb;
          const uint4* ap = (const uint4*)(aL + (size_t)kt2 * 256);
          const uint4* bp = (const uint4*)(bbase + (size_t)(q * KCHUNK + kt2) * 256);
          fa.u[0] = ap[0]; fa.u[1] = ap[1];           // ds_load_b128 x2
          fb.u[0] = bp[0]; fb.u[1] = bp[1];           // global_load_b128 x2
          c = __builtin_amdgcn_wmma_f32_16x16x32_bf16(false, fa.v, false, fb.v,
                                                      (short)0, c, false, false);
        }
      }
      if (s < kT) {  // add input projection x[:,s,:] @ W_in^T (tiny, direct loads)
        const uint32_t* xab = xapack  + ((size_t)bt * kT + s) * NKI * 256 + lane * 8;
        const uint32_t* wib = winpack + (size_t)n16 * NKI * 256 + lane * 8;
#pragma unroll 2
        for (int ktile = 0; ktile < NKI; ++ktile) {
          union { uint4 u[2]; v16bf v; } fa, fb;
          const uint4* ap = (const uint4*)(xab + (size_t)ktile * 256);
          const uint4* bp = (const uint4*)(wib + (size_t)ktile * 256);
          fa.u[0] = ap[0]; fa.u[1] = ap[1];
          fb.u[0] = bp[0]; fb.u[1] = bp[1];
          c = __builtin_amdgcn_wmma_f32_16x16x32_bf16(false, fa.v, false, fb.v,
                                                      (short)0, c, false, false);
        }
      }
    }
    {
      const int ccol = wave * 16 + l16;
      const float add = (s < kT) ? bi : 0.f;
#pragma unroll
      for (int i = 0; i < 8; i++) cur[i + 8 * hl][ccol] = c[i] + add;
    }
    __syncthreads();
    if (tid < 16) {  // per-row partial sums over this 128-wide slice
      float ssum = 0.f, ssq = 0.f;
      for (int j = 0; j < NTILE; j++) { float v = cur[tid][j]; ssum += v; ssq += v * v; }
      float* rp = red + (((size_t)bt * NNS + ns) * 16 + tid) * 2;
      rp[0] = ssum; rp[1] = ssq;
    }
    grid_barrier(bars + (size_t)(s * 2 + 0) * 2, tid);

    // ---------------- phase B: full-row layernorm + pulse + repack ----------------
    if (tid < 16) {
      float ssum = 0.f, ssq = 0.f;
      for (int q = 0; q < NNS; q++) {
        const float* rp = red + (((size_t)bt * NNS + q) * 16 + tid) * 2;
        ssum += rp[0]; ssq += rp[1];
      }
      float mu  = ssum * (1.f / kN);
      float var = ssq * (1.f / kN) - mu * mu;
      mu_s[tid] = mu; rs_s[tid] = rsqrtf(var + kLN_EPS);
    }
    __syncthreads();
    {
      float mu = mu_s[prow], rs = rs_s[prow];
#pragma unroll
      for (int j = 0; j < 8; j++) {
        int cl = pcg * 8 + j;
        float v     = (cur[prow][cl] - mu) * rs * g8[j] + be8[j];
        float p     = pot[prow][cl] * kLEAK + v + in8[j];
        float gated = p - th8[j];
        float gate  = 1.f / (1.f + __expf(-gated * sa8[j]));
        float sg    = 1.f / (1.f + __expf(-gated));
        float out   = gated * sg * gate;            // silu(gated) * gate
        pot[prow][cl] = p - gated * gate;           // detached soft reset
        spk[prow][cl] = out;
      }
    }
    __syncthreads();
    if (tid < 128) {  // repack spikes slice into bf16 A fragments (next step's K)
      int ktl = tid >> 5;                 // local ktile 0..3
      int pl = tid & 31, ph = pl >> 4, pr = pl & 15;
      uint32_t d[8];
#pragma unroll
      for (int p = 0; p < 8; p++) {
        int k = ktl * KW + kidx(p, ph);
        d[p] = packbf(spk[pr][k], spk[pr][k + 1]);
      }
      uint32_t* dst = wbuf + (((size_t)bt * NKN + (ns * NKI + ktl)) * 32 + pl) * 8;
      *(uint4*)dst = *(uint4*)d; *(uint4*)(dst + 4) = *(uint4*)(d + 4);
    }
    grid_barrier(bars + (size_t)(s * 2 + 1) * 2, tid);
  }

  // ---------------- classifier partials (deterministic, no atomics) ----------------
  if (tid < 16 * kC) {
    int row = tid & 15, cc = tid >> 4;
    const float* wr = W_cls + (size_t)cc * kN + ns * NTILE;
    float acc = 0.f;
    for (int k = 0; k < NTILE; k++) acc += spk[row][k] * wr[k];
    cls_part[(((size_t)bt * NNS + ns) * 16 + row) * 16 + cc] = acc;
  }
}

__global__ void finalize_kernel(const float* __restrict__ cls_part,
                                const float* __restrict__ b_cls,
                                float* __restrict__ out) {
  int idx = blockIdx.x * blockDim.x + threadIdx.x;
  if (idx >= kB * kC) return;
  int b = idx / kC, cc = idx % kC;
  int bt = b >> 4, row = b & 15;
  float acc = b_cls[cc];
  for (int q = 0; q < NNS; q++)
    acc += cls_part[(((size_t)bt * NNS + q) * 16 + row) * 16 + cc];
  out[idx] = acc;
}

extern "C" void kernel_launch(void* const* d_in, const int* in_sizes, int n_in,
                              void* d_out, int out_size, void* d_ws, size_t ws_size,
                              hipStream_t stream) {
  (void)in_sizes; (void)n_in; (void)out_size; (void)ws_size;
  const float* x     = (const float*)d_in[0];
  const float* W_in  = (const float*)d_in[1];
  const float* b_in  = (const float*)d_in[2];
  const float* W_rec = (const float*)d_in[3];
  const float* gamma = (const float*)d_in[4];
  const float* beta  = (const float*)d_in[5];
  const float* thr   = (const float*)d_in[6];
  const float* intr  = (const float*)d_in[7];
  const float* steep = (const float*)d_in[8];
  const float* W_cls = (const float*)d_in[9];
  const float* b_cls = (const float*)d_in[10];
  float* out = (float*)d_out;

  char* ws = (char*)d_ws;
  size_t off = 0;
  auto alloc = [&](size_t bytes) {
    void* p = ws + off; off += (bytes + 255) & ~(size_t)255; return p;
  };
  uint32_t* wpack   = (uint32_t*)alloc((size_t)NWF * 1024);   // 8 MiB
  uint32_t* winpack = (uint32_t*)alloc((size_t)NWI * 1024);   // 512 KiB
  uint32_t* xapack  = (uint32_t*)alloc((size_t)NXA * 1024);   // 4 MiB
  uint32_t* spk0    = (uint32_t*)alloc((size_t)1 << 20);      // 1 MiB
  uint32_t* spk1    = (uint32_t*)alloc((size_t)1 << 20);      // 1 MiB
  float*    red     = (float*)alloc((size_t)NBT * NNS * 16 * 2 * 4);
  float*    clsp    = (float*)alloc((size_t)NBT * NNS * 16 * 16 * 4);
  uint32_t* bars    = (uint32_t*)alloc(4096);

  zero_kernel<<<(262144 + 255) / 256, 256, 0, stream>>>(spk0, 262144);
  zero_kernel<<<4, 256, 0, stream>>>(bars, 1024);

  constexpr int packThreads = (NWF + NWI + NXA) * 32;
  pack_kernel<<<packThreads / 256, 256, 0, stream>>>(x, W_in, W_rec, wpack, winpack, xapack);

  resonator_kernel<<<NWG, 256, 0, stream>>>(b_in, gamma, beta, thr, intr, steep, W_cls,
                                            wpack, winpack, xapack, spk0, spk1,
                                            red, clsp, bars);

  finalize_kernel<<<(kB * kC + 255) / 256, 256, 0, stream>>>(clsp, b_cls, out);
}
